// QuantizedTransformerEncoder_17085379903682
// MI455X (gfx1250) — matrix operand order repaired and checked by reading, not compiled, not run on hardware
//
#include <hip/hip_runtime.h>
#include <hip/hip_bf16.h>
#include <math.h>

#define B_   8
#define N_   512
#define C_   1024
#define H_   16
#define HD_  64
#define L_   6
#define FF_  4096
#define OUT_ 2
#define M_   (B_ * N_)      // 4096 total rows
#define LDP  40             // GEMM LDS row stride (32 bf16 row + 16B pad = 80B)
#define LDPQ 72             // attention LDS row stride (64 bf16 row + 16B pad = 144B)

typedef __attribute__((ext_vector_type(16))) __bf16 v16bf;
typedef __attribute__((ext_vector_type(8)))  float  v8f;

union BF16Frag {
    v16bf v;
    unsigned short u[16];
};

__device__ __forceinline__ unsigned short f2bf(float f) {
    unsigned int u = __float_as_uint(f);
    u += 0x7FFFu + ((u >> 16) & 1u);   // round-to-nearest-even
    return (unsigned short)(u >> 16);
}

// ---------------------------------------------------------------------------
// Tensor Data Mover: 2D tile load Global->LDS with LDS padding.
// D# built per CDNA5 ISA (group0 128b, group1 256b).
// Toolchain here exposes the 6-arg builtin:
//   (uint32x4 g0, int32x8 g1, int32x4 g2, int32x4 g3, int32x8, i32 cpol)
// padIntCode: DWORDs-per-pad-interval code (3 = 16 DW = 64B, 4 = 32 DW = 128B)
// pad amount fixed at code 3 (= 4 DWORDs = 16B) to match LDP/LDPQ strides.
// ---------------------------------------------------------------------------
#if __has_builtin(__builtin_amdgcn_tensor_load_to_lds) && __has_builtin(__builtin_amdgcn_s_wait_tensorcnt)
#define USE_TDM 1
typedef __attribute__((ext_vector_type(4))) unsigned int v4u;
typedef __attribute__((ext_vector_type(8))) int v8i_;
typedef __attribute__((ext_vector_type(4))) int v4i_;

__device__ __forceinline__ void tdm_load_2d(
    const unsigned short* gptr, unsigned lds_off, int strideElems,
    int tensorRows, int tileRows, int tileCols, int padIntCode)
{
    unsigned long long ga = (unsigned long long)(size_t)gptr;
    v4u g0;
    g0[0] = 1u;                                       // count=1, user descriptor
    g0[1] = lds_off;                                  // lds_addr (bytes)
    g0[2] = (unsigned)(ga & 0xFFFFFFFFu);             // global_addr[31:0]
    g0[3] = (unsigned)((ga >> 32) & 0x01FFFFFFu) | (2u << 30);  // addr[56:32], type=2
    v8i_ g1;
    g1[0] = (int)((1u << 16)                          // data_size = 2B (bf16)
                | (1u << 20)                          // pad_enable
                | ((unsigned)padIntCode << 22)        // pad_interval
                | (3u << 25));                        // pad_amount = 4 DWORDs (16B)
    g1[1] = (int)(((unsigned)strideElems & 0xFFFFu) << 16);            // tensor_dim0[15:0]
    g1[2] = (int)(((unsigned)strideElems >> 16)
                | (((unsigned)tensorRows & 0xFFFFu) << 16));           // dim0 hi | dim1 lo
    g1[3] = (int)(((unsigned)tensorRows >> 16)
                | ((unsigned)tileCols << 16));                         // dim1 hi | tile_dim0
    g1[4] = tileRows;                                  // tile_dim1 (tile_dim2 = 0)
    g1[5] = strideElems;                               // tensor_dim0_stride[31:0]
    g1[6] = 0;
    g1[7] = 0;
    v4i_ z4; z4[0] = 0; z4[1] = 0; z4[2] = 0; z4[3] = 0;
    v8i_ z8;
    for (int i = 0; i < 8; i++) z8[i] = 0;
    __builtin_amdgcn_tensor_load_to_lds(g0, g1, z4, z4, z8, 0);
}
#else
#define USE_TDM 0
#endif

__device__ __forceinline__ unsigned lds_offset_of(const void* p) {
    return (unsigned)(unsigned long long)(size_t)p;    // low 32 bits of shared aperture = LDS offset
}

// ---------------------------------------------------------------------------
// fp32 -> bf16 conversion
// ---------------------------------------------------------------------------
__global__ void cvt_kernel(const float* __restrict__ in, unsigned short* __restrict__ out, int n) {
    int i = blockIdx.x * blockDim.x + threadIdx.x;
    int stride = gridDim.x * blockDim.x;
    for (; i < n; i += stride) out[i] = f2bf(in[i]);
}

// ---------------------------------------------------------------------------
// WMMA GEMM: out[m,n] = epilogue( sum_k A_bf16[m,k]*W_bf16[n,k] + bias[n] )
// Block 128x128 tile, 8 waves (wave tile 32x64), K chunks of 32.
// TDM path: double-buffered tensor_load_to_lds issued by wave 0.
// EPI: 0=f32 store, 1=bf16 store, 2=GELU->bf16, 3=residual add->f32
// ---------------------------------------------------------------------------
template <int EPI>
__global__ __launch_bounds__(256) void gemm_wmma(
    const unsigned short* __restrict__ A, const unsigned short* __restrict__ W,
    const float* __restrict__ bias, float* __restrict__ outF,
    unsigned short* __restrict__ outB, const float* __restrict__ resid,
    int K, int Nout)
{
    __shared__ __align__(16) unsigned short ldsA[2][128 * LDP];
    __shared__ __align__(16) unsigned short ldsB[2][128 * LDP];

    const int tid  = threadIdx.x;
    const int lane = tid & 31;
    const int wave = tid >> 5;
    const int half = lane >> 4;
    const int l16  = lane & 15;
    const int wm   = wave & 3;
    const int wn   = wave >> 2;
    const int rowBase = blockIdx.y * 128;
    const int colBase = blockIdx.x * 128;

    v8f acc[2][4];
    for (int i = 0; i < 2; i++)
        for (int j = 0; j < 4; j++)
            for (int v = 0; v < 8; v++) acc[i][j][v] = 0.0f;

    const int nt = K / 32;

#if USE_TDM
    if (wave == 0) {
        tdm_load_2d(A + (size_t)rowBase * K, lds_offset_of(&ldsA[0][0]), K, M_, 128, 32, 3);
        tdm_load_2d(W + (size_t)colBase * K, lds_offset_of(&ldsB[0][0]), K, M_, 128, 32, 3);
    }
#endif

    for (int it = 0; it < nt; it++) {
        const int buf = it & 1;
#if USE_TDM
        if (wave == 0) {
            if (it + 1 < nt) {
                int k0n = (it + 1) * 32;
                tdm_load_2d(A + (size_t)rowBase * K + k0n, lds_offset_of(&ldsA[(it + 1) & 1][0]),
                            K, M_, 128, 32, 3);
                tdm_load_2d(W + (size_t)colBase * K + k0n, lds_offset_of(&ldsB[(it + 1) & 1][0]),
                            K, M_, 128, 32, 3);
                __builtin_amdgcn_s_wait_tensorcnt(2);   // tile `it` complete (in-order)
            } else {
                __builtin_amdgcn_s_wait_tensorcnt(0);
            }
        }
        __syncthreads();
#else
        {
            int k0 = it * 32;
            for (int i = 0; i < 2; i++) {
                int t = tid + i * 256;
                int r = t >> 2, seg = t & 3;
                *(uint4*)&ldsA[buf][r * LDP + seg * 8] =
                    *(const uint4*)(A + (size_t)(rowBase + r) * K + k0 + seg * 8);
                *(uint4*)&ldsB[buf][r * LDP + seg * 8] =
                    *(const uint4*)(W + (size_t)(colBase + r) * K + k0 + seg * 8);
            }
            if (k0 + 32 < K) {
                int r = tid >> 1;
                __builtin_prefetch(A + (size_t)(rowBase + r) * K + k0 + 32, 0, 0);
                __builtin_prefetch(W + (size_t)(colBase + r) * K + k0 + 32, 0, 0);
            }
            __syncthreads();
        }
#endif
        BF16Frag af[2], bf[4];
        for (int tm = 0; tm < 2; tm++) {
            int m = wm * 32 + tm * 16 + l16;
            for (int v = 0; v < 8; v++) {
                int kk = (v & 3) * 2 + (v >> 2) * 16 + half * 8;   // A 16x32 layout
                af[tm].u[2 * v]     = ldsA[buf][m * LDP + kk];
                af[tm].u[2 * v + 1] = ldsA[buf][m * LDP + kk + 1];
            }
        }
        for (int tn = 0; tn < 4; tn++) {
            int n = wn * 64 + tn * 16 + l16;
            for (int v = 0; v < 8; v++) {
                int kk = (v & 3) * 2 + (v >> 2) * 16 + half * 8;   // B 32x16 layout
                bf[tn].u[2 * v]     = ldsB[buf][n * LDP + kk];
                bf[tn].u[2 * v + 1] = ldsB[buf][n * LDP + kk + 1];
            }
        }
        for (int tm = 0; tm < 2; tm++)
            for (int tn = 0; tn < 4; tn++)
                acc[tm][tn] = __builtin_amdgcn_wmma_f32_16x16x32_bf16(
                    false, af[tm].v, false, bf[tn].v, (short)0, acc[tm][tn], false, false);
        __syncthreads();
    }

    for (int tm = 0; tm < 2; tm++) {
        for (int tn = 0; tn < 4; tn++) {
            int nG = colBase + wn * 64 + tn * 16 + l16;
            float bval = bias[nG];
            for (int v = 0; v < 8; v++) {
                int mG = rowBase + wm * 32 + tm * 16 + v + half * 8;
                float val = acc[tm][tn][v] + bval;
                size_t idx = (size_t)mG * Nout + nG;
                if (EPI == 0) {
                    outF[idx] = val;
                } else if (EPI == 1) {
                    outB[idx] = f2bf(val);
                } else if (EPI == 2) {
                    float g = 0.5f * val * (1.0f + erff(val * 0.70710678118654752f));
                    outB[idx] = f2bf(g);
                } else {
                    outF[idx] = resid[idx] + val;
                }
            }
        }
    }
}

// ---------------------------------------------------------------------------
// LayerNorm: one block per row of 1024, writes bf16 normalized row
// ---------------------------------------------------------------------------
__global__ __launch_bounds__(256) void ln_kernel(
    const float* __restrict__ x, const float* __restrict__ g,
    const float* __restrict__ b, unsigned short* __restrict__ hout)
{
    __shared__ float red[256];
    const int row = blockIdx.x, tid = threadIdx.x;
    const float* xr = x + (size_t)row * C_;
    float v4[4], s = 0.f;
    for (int i = 0; i < 4; i++) { v4[i] = xr[tid + i * 256]; s += v4[i]; }
    red[tid] = s; __syncthreads();
    for (int o = 128; o > 0; o >>= 1) { if (tid < o) red[tid] += red[tid + o]; __syncthreads(); }
    float mu = red[0] * (1.0f / C_);
    __syncthreads();
    float vs = 0.f;
    for (int i = 0; i < 4; i++) { float d = v4[i] - mu; vs += d * d; }
    red[tid] = vs; __syncthreads();
    for (int o = 128; o > 0; o >>= 1) { if (tid < o) red[tid] += red[tid + o]; __syncthreads(); }
    float rstd = rsqrtf(red[0] * (1.0f / C_) + 1e-6f);
    for (int i = 0; i < 4; i++) {
        int c = tid + i * 256;
        hout[(size_t)row * C_ + c] = f2bf((v4[i] - mu) * rstd * g[c] + b[c]);
    }
}

// ---------------------------------------------------------------------------
// Flash attention: block = (b, h, 128 q rows); 8 waves x 16 q rows.
// KV chunks of 32 through LDS (TDM when available). Softmax denominator is
// accumulated by a 9th WMMA against a ones-column B fragment.
// ---------------------------------------------------------------------------
__global__ __launch_bounds__(256) void attn_kernel(
    const unsigned short* __restrict__ qkv, const int* __restrict__ mask,
    unsigned short* __restrict__ o_out)
{
    __shared__ __align__(16) unsigned short ldsQ[128 * LDPQ];
    __shared__ __align__(16) unsigned short ldsK[32 * LDPQ];
    __shared__ __align__(16) unsigned short ldsV[32 * LDPQ];
    __shared__ __align__(16) unsigned short ldsP[8 * 16 * 36];

    const int tid  = threadIdx.x;
    const int lane = tid & 31;
    const int wave = tid >> 5;
    const int half = lane >> 4;
    const int l16  = lane & 15;
    const int b = blockIdx.z, h = blockIdx.y, qBase = blockIdx.x * 128;
    const int rowStride = 3 * C_;

    const unsigned short* qptr  = qkv + (size_t)(b * N_ + qBase) * rowStride + h * HD_;
    const unsigned short* kbase = qkv + (size_t)(b * N_) * rowStride + C_ + h * HD_;
    const unsigned short* vbase = qkv + (size_t)(b * N_) * rowStride + 2 * C_ + h * HD_;

    // ---- stage Q (128 x 64) ----
#if USE_TDM
    if (wave == 0) {
        tdm_load_2d(qptr, lds_offset_of(&ldsQ[0]), rowStride, N_, 128, 64, 4);
        __builtin_amdgcn_s_wait_tensorcnt(0);
    }
    __syncthreads();
#else
    for (int i = 0; i < 4; i++) {
        int t = tid + i * 256;
        int r = t >> 3, seg = t & 7;
        *(uint4*)&ldsQ[r * LDPQ + seg * 8] = *(const uint4*)(qptr + (size_t)r * rowStride + seg * 8);
    }
    __syncthreads();
#endif

    BF16Frag qf[2];
    {
        int m = wave * 16 + l16;
        for (int c = 0; c < 2; c++)
            for (int v = 0; v < 8; v++) {
                int kk = (v & 3) * 2 + (v >> 2) * 16 + half * 8 + c * 32;
                qf[c].u[2 * v]     = ldsQ[m * LDPQ + kk];
                qf[c].u[2 * v + 1] = ldsQ[m * LDPQ + kk + 1];
            }
    }

    BF16Frag ones;      // B fragment: column 0 all 1.0, other columns 0
    for (int j = 0; j < 16; j++) ones.u[j] = (l16 == 0) ? 0x3F80 : 0;

    float mrun[8], alpha[8];
    v8f ofr[4], lacc;
    for (int v = 0; v < 8; v++) { mrun[v] = -1e30f; lacc[v] = 0.f; }
    for (int f = 0; f < 4; f++)
        for (int v = 0; v < 8; v++) ofr[f][v] = 0.f;
    const float scale = 0.125f;    // HD^-0.5
    unsigned short* myP = &ldsP[wave * 16 * 36];

    for (int jc = 0; jc < N_ / 32; jc++) {
#if USE_TDM
        if (wave == 0) {
            tdm_load_2d(kbase + (size_t)(jc * 32) * rowStride, lds_offset_of(&ldsK[0]),
                        rowStride, N_, 32, 64, 4);
            tdm_load_2d(vbase + (size_t)(jc * 32) * rowStride, lds_offset_of(&ldsV[0]),
                        rowStride, N_, 32, 64, 4);
            __builtin_amdgcn_s_wait_tensorcnt(0);
        }
        __syncthreads();
#else
        {
            int r = tid >> 3, seg = tid & 7;
            *(uint4*)&ldsK[r * LDPQ + seg * 8] =
                *(const uint4*)(kbase + (size_t)(jc * 32 + r) * rowStride + seg * 8);
            *(uint4*)&ldsV[r * LDPQ + seg * 8] =
                *(const uint4*)(vbase + (size_t)(jc * 32 + r) * rowStride + seg * 8);
        }
        __syncthreads();
#endif

        // S tiles: cols jc*32..+15 (s0), +16..+31 (s1); K-dim = HD = 64 (2 chunks)
        v8f s0, s1;
        for (int v = 0; v < 8; v++) { s0[v] = 0.f; s1[v] = 0.f; }
        for (int c = 0; c < 2; c++) {
            BF16Frag bk0, bk1;
            for (int v = 0; v < 8; v++) {
                int kk = (v & 3) * 2 + (v >> 2) * 16 + half * 8 + c * 32;          // head-dim index
                bk0.u[2 * v]     = ldsK[l16 * LDPQ + kk];
                bk0.u[2 * v + 1] = ldsK[l16 * LDPQ + kk + 1];
                bk1.u[2 * v]     = ldsK[(16 + l16) * LDPQ + kk];
                bk1.u[2 * v + 1] = ldsK[(16 + l16) * LDPQ + kk + 1];
            }
            s0 = __builtin_amdgcn_wmma_f32_16x16x32_bf16(false, qf[c].v, false, bk0.v, (short)0, s0, false, false);
            s1 = __builtin_amdgcn_wmma_f32_16x16x32_bf16(false, qf[c].v, false, bk1.v, (short)0, s1, false, false);
        }

        float mb0 = mask[b * N_ + jc * 32 + l16]      ? 0.f : -1e9f;
        float mb1 = mask[b * N_ + jc * 32 + 16 + l16] ? 0.f : -1e9f;

        float p0[8], p1[8];
        for (int v = 0; v < 8; v++) {
            float a0 = s0[v] * scale + mb0;
            float a1 = s1[v] * scale + mb1;
            float mt = fmaxf(a0, a1);
            for (int off = 1; off < 16; off <<= 1) mt = fmaxf(mt, __shfl_xor(mt, off, 32));
            float mn = fmaxf(mrun[v], mt);
            float al = expf(mrun[v] - mn);
            mrun[v] = mn; alpha[v] = al;
            p0[v] = expf(a0 - mn);
            p1[v] = expf(a1 - mn);
        }
        for (int f = 0; f < 4; f++)
            for (int v = 0; v < 8; v++) ofr[f][v] *= alpha[v];
        for (int v = 0; v < 8; v++) lacc[v] *= alpha[v];

        // stage P (16x32) to re-layout as an A fragment
        for (int v = 0; v < 8; v++) {
            int r = v + half * 8;
            myP[r * 36 + l16]      = f2bf(p0[v]);
            myP[r * 36 + 16 + l16] = f2bf(p1[v]);
        }
        BF16Frag pf;
        for (int v = 0; v < 8; v++) {
            int kk = (v & 3) * 2 + (v >> 2) * 16 + half * 8;
            pf.u[2 * v]     = myP[l16 * 36 + kk];
            pf.u[2 * v + 1] = myP[l16 * 36 + kk + 1];
        }
        // O += P (16x32) x V (32x64); denominator rides a 9th WMMA
        for (int f = 0; f < 4; f++) {
            BF16Frag vf;
            for (int v = 0; v < 8; v++) {
                int kk = (v & 3) * 2 + (v >> 2) * 16 + half * 8;   // kv row
                vf.u[2 * v]     = ldsV[kk * LDPQ + f * 16 + l16];
                vf.u[2 * v + 1] = ldsV[(kk + 1) * LDPQ + f * 16 + l16];
            }
            ofr[f] = __builtin_amdgcn_wmma_f32_16x16x32_bf16(false, pf.v, false, vf.v, (short)0, ofr[f], false, false);
        }
        lacc = __builtin_amdgcn_wmma_f32_16x16x32_bf16(false, pf.v, false, ones.v, (short)0, lacc, false, false);
        __syncthreads();
    }

    for (int f = 0; f < 4; f++) {
        int d = f * 16 + l16;
        for (int v = 0; v < 8; v++) {
            int q = qBase + wave * 16 + v + half * 8;
            float lsum = __shfl(lacc[v], half * 16, 32);      // column 0 holds row sum
            float val = ofr[f][v] / lsum;
            o_out[(size_t)(b * N_ + q) * C_ + h * HD_ + d] = f2bf(val);
        }
    }
}

// ---------------------------------------------------------------------------
__global__ void addpos_kernel(float* __restrict__ x, const float* __restrict__ pos) {
    size_t i = (size_t)blockIdx.x * blockDim.x + threadIdx.x;
    x[i] += pos[i & ((size_t)N_ * C_ - 1)];
}

// ---------------------------------------------------------------------------
__global__ __launch_bounds__(256) void out_kernel(
    const float* __restrict__ x, const float* __restrict__ Wout,
    const float* __restrict__ bout, float* __restrict__ out)
{
    int tid = threadIdx.x, lane = tid & 31, wave = tid >> 5;
    int row = blockIdx.x * 8 + wave;
    const float* xr = x + (size_t)row * C_;
    float a0 = 0.f, a1 = 0.f;
    for (int c = lane; c < C_; c += 32) {
        float xv = xr[c];
        a0 += xv * Wout[c];
        a1 += xv * Wout[C_ + c];
    }
    for (int off = 16; off > 0; off >>= 1) {
        a0 += __shfl_xor(a0, off, 32);
        a1 += __shfl_xor(a1, off, 32);
    }
    if (lane == 0) {
        out[row * 2 + 0] = 1.f / (1.f + expf(-(a0 + bout[0])));
        out[row * 2 + 1] = 1.f / (1.f + expf(-(a1 + bout[1])));
    }
}

// ---------------------------------------------------------------------------
extern "C" void kernel_launch(void* const* d_in, const int* in_sizes, int n_in,
                              void* d_out, int out_size, void* d_ws, size_t ws_size,
                              hipStream_t stream) {
    const float* inputs = (const float*)d_in[0];
    const int*   amask  = (const int*)d_in[1];
    const float* pos    = (const float*)d_in[2];
    const float* Wi     = (const float*)d_in[3];
    const float* bi     = (const float*)d_in[4];
    const float* ln1g   = (const float*)d_in[5];
    const float* ln1b   = (const float*)d_in[6];
    const float* Wqkv   = (const float*)d_in[7];
    const float* bqkv   = (const float*)d_in[8];
    const float* Wo     = (const float*)d_in[9];
    const float* bo     = (const float*)d_in[10];
    const float* ln2g   = (const float*)d_in[11];
    const float* ln2b   = (const float*)d_in[12];
    const float* W1     = (const float*)d_in[13];
    const float* b1     = (const float*)d_in[14];
    const float* W2     = (const float*)d_in[15];
    const float* b2     = (const float*)d_in[16];
    const float* Wout   = (const float*)d_in[17];
    const float* bout   = (const float*)d_in[18];
    float* out = (float*)d_out;

    char* ws = (char*)d_ws;
    size_t off = 0;
    auto allocU = [&](size_t n) { unsigned short* p = (unsigned short*)(ws + off); off += n * 2; return p; };
    auto allocF = [&](size_t n) { float* p = (float*)(ws + off); off += n * 4; return p; };

    unsigned short* Wi_bf   = allocU((size_t)C_ * C_);
    unsigned short* Wqkv_bf = allocU((size_t)L_ * 3 * C_ * C_);
    unsigned short* Wo_bf   = allocU((size_t)L_ * C_ * C_);
    unsigned short* W1_bf   = allocU((size_t)L_ * FF_ * C_);
    unsigned short* W2_bf   = allocU((size_t)L_ * C_ * FF_);
    unsigned short* in_bf   = allocU((size_t)M_ * C_);
    unsigned short* h_bf    = allocU((size_t)M_ * C_);
    unsigned short* qkv_bf  = allocU((size_t)M_ * 3 * C_);
    unsigned short* o_bf    = allocU((size_t)M_ * C_);
    unsigned short* f_bf    = allocU((size_t)M_ * FF_);
    float*          xbuf    = allocF((size_t)M_ * C_);

    dim3 blk(256);
    auto cvt = [&](const float* src, unsigned short* dst, int n) {
        cvt_kernel<<<4096, blk, 0, stream>>>(src, dst, n);
    };
    cvt(inputs, in_bf, M_ * C_);
    cvt(Wi, Wi_bf, C_ * C_);
    cvt(Wqkv, Wqkv_bf, L_ * 3 * C_ * C_);
    cvt(Wo, Wo_bf, L_ * C_ * C_);
    cvt(W1, W1_bf, L_ * FF_ * C_);
    cvt(W2, W2_bf, L_ * C_ * FF_);

    gemm_wmma<0><<<dim3(C_ / 128, M_ / 128), blk, 0, stream>>>(
        in_bf, Wi_bf, bi, xbuf, nullptr, nullptr, C_, C_);
    addpos_kernel<<<(M_ * C_) / 256, blk, 0, stream>>>(xbuf, pos);

    for (int l = 0; l < L_; l++) {
        ln_kernel<<<M_, blk, 0, stream>>>(xbuf, ln1g + l * C_, ln1b + l * C_, h_bf);
        gemm_wmma<1><<<dim3(3 * C_ / 128, M_ / 128), blk, 0, stream>>>(
            h_bf, Wqkv_bf + (size_t)l * 3 * C_ * C_, bqkv + l * 3 * C_,
            nullptr, qkv_bf, nullptr, C_, 3 * C_);
        attn_kernel<<<dim3(N_ / 128, H_, B_), blk, 0, stream>>>(qkv_bf, amask, o_bf);
        gemm_wmma<3><<<dim3(C_ / 128, M_ / 128), blk, 0, stream>>>(
            o_bf, Wo_bf + (size_t)l * C_ * C_, bo + l * C_,
            xbuf, nullptr, xbuf, C_, C_);
        ln_kernel<<<M_, blk, 0, stream>>>(xbuf, ln2g + l * C_, ln2b + l * C_, h_bf);
        gemm_wmma<2><<<dim3(FF_ / 128, M_ / 128), blk, 0, stream>>>(
            h_bf, W1_bf + (size_t)l * FF_ * C_, b1 + l * FF_,
            nullptr, f_bf, nullptr, C_, FF_);
        gemm_wmma<3><<<dim3(C_ / 128, M_ / 128), blk, 0, stream>>>(
            f_bf, W2_bf + (size_t)l * C_ * FF_, b2 + l * C_,
            xbuf, nullptr, xbuf, FF_, C_);
    }

    out_kernel<<<M_ / 8, blk, 0, stream>>>(xbuf, Wout, bout, out);
}
